// Head_30013231464535
// MI455X (gfx1250) — matrix operand order repaired and checked by reading, not compiled, not run on hardware
//
#include <hip/hip_runtime.h>

typedef __attribute__((ext_vector_type(16))) __bf16 v16bf;
typedef __attribute__((ext_vector_type(8)))  float  v8f;
typedef int v4i __attribute__((vector_size(16)));

#define BB 8
#define TT 2048
#define DD 1024

#define __AS(n) __attribute__((address_space(n)))

#if defined(__has_builtin)
#if __has_builtin(__builtin_amdgcn_global_load_async_to_lds_b128)
#define HAVE_ASYNC_LDS 1
#endif
#endif

union Frag16 { v16bf v; uint4 q[2]; };

// A-matrix 16x32 bf16 fragment: per-lane run k = h*16+[0..15]
__device__ __forceinline__ v16bf load_a_frag(const __bf16* p, int h) {
    Frag16 f;
    f.q[0] = *reinterpret_cast<const uint4*>(p + h * 16);
    f.q[1] = *reinterpret_cast<const uint4*>(p + h * 16 + 8);
    return f.v;
}
// B-matrix 32x16 bf16 fragment: per-lane run k = h*16+[0..15] (caller adds h*16)
__device__ __forceinline__ v16bf load_b_frag(const __bf16* p) {
    Frag16 f;
    f.q[0] = *reinterpret_cast<const uint4*>(p);
    f.q[1] = *reinterpret_cast<const uint4*>(p + 8);
    return f.v;
}

__device__ __forceinline__ void cp16_to_lds(const __bf16* g, __bf16* l) {
#ifdef HAVE_ASYNC_LDS
    __builtin_amdgcn_global_load_async_to_lds_b128(
        (__AS(1) v4i*)g, (__AS(3) v4i*)l, 0, 0);
#else
    *reinterpret_cast<uint4*>(l) = *reinterpret_cast<const uint4*>(g);
#endif
}
__device__ __forceinline__ void async_wait0() {
#ifdef HAVE_ASYNC_LDS
#if __has_builtin(__builtin_amdgcn_s_wait_asynccnt)
    __builtin_amdgcn_s_wait_asynccnt(0);
#else
    asm volatile("s_wait_asynccnt 0x0" ::: "memory");
#endif
#endif
}

// ---------------------------------------------------------------------------
// Stage 0a: fp32 -> bf16 (x)
// ---------------------------------------------------------------------------
__global__ void cvt_bf16(const float* __restrict__ in, __bf16* __restrict__ out, int n) {
    int i = blockIdx.x * blockDim.x + threadIdx.x;
    int stride = gridDim.x * blockDim.x;
    for (; i < n; i += stride) out[i] = (__bf16)in[i];
}
// Stage 0b: fp32 -> bf16 with transpose (weights): out[n*D+k] = in[k*D+n]
__global__ void cvtT_bf16(const float* __restrict__ in, __bf16* __restrict__ out, int n) {
    int i = blockIdx.x * blockDim.x + threadIdx.x;
    int stride = gridDim.x * blockDim.x;
    for (; i < n; i += stride) {
        int nn = i >> 10, k = i & (DD - 1);
        out[i] = (__bf16)in[k * DD + nn];
    }
}

// ---------------------------------------------------------------------------
// Stage 1: QKV projection GEMM.  M=16384, N=1024, K=1024, Wt pre-transposed.
// 128x128 tile / block, 8 waves x (16 rows x 128 cols).  Weight tile double-
// buffered in LDS via async global->LDS copies.  V output stored transposed.
// ---------------------------------------------------------------------------
__global__ __launch_bounds__(256) void proj_gemm(
        const __bf16* __restrict__ X,
        const __bf16* __restrict__ WtK, const __bf16* __restrict__ WtQ, const __bf16* __restrict__ WtV,
        __bf16* __restrict__ Ko, __bf16* __restrict__ Qo, __bf16* __restrict__ Vt) {
    __shared__ __bf16 sB[2][128 * 32];   // [n][k] tile, fragment-contiguous

    const __bf16* Wt = (blockIdx.z == 0) ? WtK : (blockIdx.z == 1) ? WtQ : WtV;

    const int t    = threadIdx.x;
    const int lane = t & 31;
    const int w    = t >> 5;
    const int h    = lane >> 4;
    const int l15  = lane & 15;
    const int m0   = blockIdx.y * 128 + w * 16;
    const int n0   = blockIdx.x * 128;

    v8f acc[8];
    #pragma unroll
    for (int i = 0; i < 8; i++)
        #pragma unroll
        for (int r = 0; r < 8; r++) acc[i][r] = 0.0f;

    const __bf16* arow = X + (size_t)(m0 + l15) * DD;

    // stage Wt[n0 + (t>>1)][kk + (t&1)*16 .. +15] (32B per thread)
    const int srow = t >> 1, scol = (t & 1) * 16;
    {
        const __bf16* src = Wt + (size_t)(n0 + srow) * DD + scol;
        __bf16* dst = &sB[0][srow * 32 + scol];
        cp16_to_lds(src, dst);
        cp16_to_lds(src + 8, dst + 8);
    }

    int cur = 0;
    for (int kk = 0; kk < DD; kk += 32) {
        async_wait0();
        __syncthreads();
        if (kk + 32 < DD) {     // issue next tile; overlaps with WMMA below
            const __bf16* src = Wt + (size_t)(n0 + srow) * DD + kk + 32 + scol;
            __bf16* dst = &sB[cur ^ 1][srow * 32 + scol];
            cp16_to_lds(src, dst);
            cp16_to_lds(src + 8, dst + 8);
        }

        // preload all 8 B fragments (one dscnt wait), then 8 back-to-back WMMAs
        v16bf bfr[8];
        #pragma unroll
        for (int nt = 0; nt < 8; nt++)
            bfr[nt] = load_b_frag(&sB[cur][(nt * 16 + l15) * 32 + h * 16]);
        v16bf a = load_a_frag(arow + kk, h);
        #pragma unroll
        for (int nt = 0; nt < 8; nt++)
            acc[nt] = __builtin_amdgcn_wmma_f32_16x16x32_bf16(
                false, a, false, bfr[nt], (short)0, acc[nt], false, false);
        cur ^= 1;
    }

    if (blockIdx.z == 2) {
        // V stored transposed: Vt[batch][d][t]
        #pragma unroll
        for (int nt = 0; nt < 8; nt++)
            #pragma unroll
            for (int r = 0; r < 8; r++) {
                int row = m0 + r + 8 * h;             // global token index
                int col = n0 + nt * 16 + l15;         // feature
                int batch = row >> 11, tok = row & (TT - 1);
                Vt[((size_t)batch * DD + col) * TT + tok] = (__bf16)acc[nt][r];
            }
    } else {
        __bf16* O = (blockIdx.z == 0) ? Ko : Qo;
        #pragma unroll
        for (int nt = 0; nt < 8; nt++)
            #pragma unroll
            for (int r = 0; r < 8; r++) {
                int row = m0 + r + 8 * h;
                int col = n0 + nt * 16 + l15;
                O[(size_t)row * DD + col] = (__bf16)acc[nt][r];
            }
    }
}

// ---------------------------------------------------------------------------
// Stage 2: flash attention.  One block = (batch, 32 query rows), 8 waves.
// S = Q*K^T per 64-key block (8 waves x one 16x16 tile, software-pipelined),
// parallel online softmax in LDS, then each wave rescales + accumulates P*V
// for its 128 output columns.
// ---------------------------------------------------------------------------
__global__ __launch_bounds__(256) void attn_fwd(
        const __bf16* __restrict__ Q, const __bf16* __restrict__ K,
        const __bf16* __restrict__ Vt, float* __restrict__ out) {
    __shared__ float  sS[32 * 64];
    __shared__ __bf16 sP[32 * 64];
    __shared__ float  sM[32], sL[32], sA[32];
    __shared__ float  sRed[256];

    const int t     = threadIdx.x;
    const int lane  = t & 31;
    const int w     = t >> 5;
    const int h     = lane >> 4;
    const int l15   = lane & 15;
    const int batch = blockIdx.y;
    const int qbase = blockIdx.x * 32;

    const __bf16* Qb = Q + (size_t)batch * TT * DD;
    const __bf16* Kb = K + (size_t)batch * TT * DD;

    v8f acc[16];                       // [mt*8+nt] : 32 rows x 128 cols slice
    #pragma unroll
    for (int i = 0; i < 16; i++)
        #pragma unroll
        for (int r = 0; r < 8; r++) acc[i][r] = 0.0f;

    if (t < 32) { sM[t] = -__builtin_inff(); sL[t] = 0.0f; }
    __syncthreads();

    const int   mt_s = w >> 2;         // S-tile row (0..1)
    const int   nt_s = w & 3;          // S-tile col (0..3)
    const float scale = 0.03125f;      // 1/sqrt(1024)
    const int   nblocks = qbase / 64 + 1;

    const __bf16* qrow = Qb + (size_t)(qbase + mt_s * 16 + l15) * DD;
    const int prow = t >> 3, pc0 = (t & 7) * 8;   // softmax row / col-chunk

    for (int j = 0; j < nblocks; j++) {
        const int kbase = j * 64;

        if (j + 1 < nblocks) {          // prefetch next key/value block
            __builtin_prefetch(Kb + (size_t)(kbase + 64 + nt_s * 16 + l15) * DD, 0, 1);
            __builtin_prefetch(Vt + ((size_t)batch * DD + w * 128 + l15) * TT + kbase + 64, 0, 1);
        }

        // ---- S tile: Q(16xD) * K^T -> 16x16 per wave, 2-stage pipeline ----
        v8f c8;
        #pragma unroll
        for (int r = 0; r < 8; r++) c8[r] = 0.0f;
        const __bf16* krow = Kb + (size_t)(kbase + nt_s * 16 + l15) * DD;
        v16bf a0 = load_a_frag(qrow, h);
        v16bf b0 = load_b_frag(krow + h * 16);
        #pragma unroll 4
        for (int d = 0; d < DD - 32; d += 32) {
            v16bf a1 = load_a_frag(qrow + d + 32, h);
            v16bf b1 = load_b_frag(krow + d + 32 + h * 16);
            c8 = __builtin_amdgcn_wmma_f32_16x16x32_bf16(
                false, a0, false, b0, (short)0, c8, false, false);
            a0 = a1; b0 = b1;
        }
        c8 = __builtin_amdgcn_wmma_f32_16x16x32_bf16(
            false, a0, false, b0, (short)0, c8, false, false);

        #pragma unroll
        for (int r = 0; r < 8; r++) {
            int row = qbase + mt_s * 16 + r + 8 * h;
            int col = kbase + nt_s * 16 + l15;
            float v = c8[r] * scale;
            if (col > row) v = -__builtin_inff();          // causal mask
            sS[(row - qbase) * 64 + (col - kbase)] = v;
        }
        __syncthreads();

        // ---- parallel online softmax ----
        {   // per-thread partial max over 8 cols
            float m8 = sS[prow * 64 + pc0];
            #pragma unroll
            for (int i = 1; i < 8; i++) m8 = fmaxf(m8, sS[prow * 64 + pc0 + i]);
            sRed[t] = m8;
        }
        __syncthreads();
        if (t < 32) {
            float mold = sM[t], mnew = mold;
            #pragma unroll
            for (int i = 0; i < 8; i++) mnew = fmaxf(mnew, sRed[t * 8 + i]);
            sA[t] = __expf(mold - mnew);
            sM[t] = mnew;
        }
        __syncthreads();
        {   // exp + bf16 pack (one b128 store) + partial sum
            float mn = sM[prow], s = 0.0f;
            union { __bf16 b[8]; uint4 q; } pk;
            #pragma unroll
            for (int i = 0; i < 8; i++) {
                float p = __expf(sS[prow * 64 + pc0 + i] - mn);
                s += p;
                pk.b[i] = (__bf16)p;
            }
            *reinterpret_cast<uint4*>(&sP[prow * 64 + pc0]) = pk.q;
            sRed[t] = s;
        }
        __syncthreads();
        if (t < 32) {
            float s = 0.0f;
            #pragma unroll
            for (int i = 0; i < 8; i++) s += sRed[t * 8 + i];
            sL[t] = sL[t] * sA[t] + s;
        }

        // ---- rescale O, accumulate P*V (wave w: output cols w*128..+127) ----
        float alph[16];
        #pragma unroll
        for (int mt = 0; mt < 2; mt++)
            #pragma unroll
            for (int r = 0; r < 8; r++) alph[mt * 8 + r] = sA[mt * 16 + r + 8 * h];
        #pragma unroll
        for (int ct = 0; ct < 16; ct++) {
            int mt = ct >> 3;
            #pragma unroll
            for (int r = 0; r < 8; r++) acc[ct][r] *= alph[mt * 8 + r];
        }
        #pragma unroll
        for (int ks = 0; ks < 2; ks++) {
            v16bf pa[2];
            #pragma unroll
            for (int mt = 0; mt < 2; mt++)
                pa[mt] = load_a_frag(&sP[(mt * 16 + l15) * 64 + ks * 32], h);
            #pragma unroll
            for (int ng = 0; ng < 2; ng++) {          // 2 groups of 4 V frags
                v16bf bv[4];
                #pragma unroll
                for (int q4 = 0; q4 < 4; q4++) {
                    int n = w * 128 + (ng * 4 + q4) * 16 + l15;
                    bv[q4] = load_b_frag(
                        Vt + ((size_t)batch * DD + n) * TT + kbase + ks * 32 + h * 16);
                }
                #pragma unroll
                for (int q4 = 0; q4 < 4; q4++)
                    #pragma unroll
                    for (int mt = 0; mt < 2; mt++)
                        acc[mt * 8 + ng * 4 + q4] = __builtin_amdgcn_wmma_f32_16x16x32_bf16(
                            false, pa[mt], false, bv[q4], (short)0,
                            acc[mt * 8 + ng * 4 + q4], false, false);
            }
        }
    }
    __syncthreads();   // sL final values visible to all

    // ---- epilogue: divide by l, store fp32 ----
    float linv[16];
    #pragma unroll
    for (int mt = 0; mt < 2; mt++)
        #pragma unroll
        for (int r = 0; r < 8; r++) linv[mt * 8 + r] = 1.0f / sL[mt * 16 + r + 8 * h];
    float* ob = out + (size_t)batch * TT * DD;
    #pragma unroll
    for (int ct = 0; ct < 16; ct++) {
        int mt = ct >> 3, nt = ct & 7;
        #pragma unroll
        for (int r = 0; r < 8; r++) {
            int row = qbase + mt * 16 + r + 8 * h;
            int col = w * 128 + nt * 16 + l15;
            ob[(size_t)row * DD + col] = acc[ct][r] * linv[mt * 8 + r];
        }
    }
}

// ---------------------------------------------------------------------------
extern "C" void kernel_launch(void* const* d_in, const int* in_sizes, int n_in,
                              void* d_out, int out_size, void* d_ws, size_t ws_size,
                              hipStream_t stream) {
    const float* x  = (const float*)d_in[0];
    const float* Wk = (const float*)d_in[1];
    const float* Wq = (const float*)d_in[2];
    const float* Wv = (const float*)d_in[3];
    float* out = (float*)d_out;

    const size_t XE = (size_t)BB * TT * DD;   // 16,777,216
    const size_t WE = (size_t)DD * DD;        //  1,048,576

    char* p = (char*)d_ws;
    __bf16* xb  = (__bf16*)p; p += XE * 2;
    __bf16* wkb = (__bf16*)p; p += WE * 2;    // transposed
    __bf16* wqb = (__bf16*)p; p += WE * 2;    // transposed
    __bf16* wvb = (__bf16*)p; p += WE * 2;    // transposed
    __bf16* kb  = (__bf16*)p; p += XE * 2;    // [b][t][d]
    __bf16* qb  = (__bf16*)p; p += XE * 2;    // [b][t][d]
    __bf16* vtb = (__bf16*)p;                 // [b][d][t]

    cvt_bf16 <<<2048, 256, 0, stream>>>(x,  xb,  (int)XE);
    cvtT_bf16<<<512,  256, 0, stream>>>(Wk, wkb, (int)WE);
    cvtT_bf16<<<512,  256, 0, stream>>>(Wq, wqb, (int)WE);
    cvtT_bf16<<<512,  256, 0, stream>>>(Wv, wvb, (int)WE);

    dim3 g1(DD / 128, (BB * TT) / 128, 3);    // (8, 128, 3)
    proj_gemm<<<g1, 256, 0, stream>>>(xb, wkb, wqb, wvb, kb, qb, vtb);

    dim3 g2(TT / 32, BB);                     // (64, 8)
    attn_fwd<<<g2, 256, 0, stream>>>(qb, kb, vtb, out);
}
